// MultiHeadAttention_43800076484692
// MI455X (gfx1250) — compile-verified, matrix operand order checked by reading, not run on hardware
//
#include <hip/hip_runtime.h>

// ---------------------------------------------------------------------------
// MultiHeadAttention for MI455X (gfx1250):
//   proj (WMMA f16, fp32 acc) x3 -> flash attention (WMMA + async global->LDS
//   double-buffered staging + LDS online softmax) -> output proj (WMMA).
// Wave32. fp16 intermediates in d_ws (4 x 8 MB).
// ---------------------------------------------------------------------------

typedef _Float16 half_t;
typedef __attribute__((ext_vector_type(16))) _Float16 v16h;
typedef __attribute__((ext_vector_type(2)))  _Float16 h2;
typedef __attribute__((ext_vector_type(8)))  float    v8f;

#define B_  4
#define T_  2048
#define D_  512
#define H_  8
#define DK  64
#define NEGV -65504.0f
#define SCALE_ 0.044194173824159216f   // D^-0.5 (reference scales by full dim)

#define WMMA(a, b, c) \
  __builtin_amdgcn_wmma_f32_16x16x32_f16(false, (a), false, (b), (short)0, (c), false, false)

// K-offset pattern inside a 16-bit A/B fragment: VGPR i, lane-group g = lane/16.
// (ISA 7.12.2: lanes 0-15 hold K={0,1},{2,3},{4,5},{6,7},{16,17}... ; lanes 16-31 same +8)
__device__ __forceinline__ int frag_k(int i, int g) {
  return ((i < 4) ? (2 * i) : (16 + 2 * (i - 4))) + 8 * g;
}

// A/B fragment (16x32, f16) from row-major fp32 source; row = M (A) or N (B).
__device__ __forceinline__ v16h load_frag_f32(const float* __restrict__ src, int ld,
                                              int row, int k0, int g) {
  v16h a;
  const float* p = src + (size_t)row * ld + k0;
#pragma unroll
  for (int i = 0; i < 8; ++i) {
    int kk = frag_k(i, g);
    float2 xy = *(const float2*)(p + kk);
    a[2 * i]     = (_Float16)xy.x;
    a[2 * i + 1] = (_Float16)xy.y;
  }
  return a;
}

// A/B fragment from row-major fp16 source (global or LDS after inlining).
__device__ __forceinline__ v16h load_frag_f16(const half_t* __restrict__ src, int ld,
                                              int row, int k0, int g) {
  v16h a;
  const half_t* p = src + (size_t)row * ld + k0;
#pragma unroll
  for (int i = 0; i < 8; ++i) {
    int kk = frag_k(i, g);
    h2 xy = *(const h2*)(p + kk);
    a[2 * i]     = xy.x;
    a[2 * i + 1] = xy.y;
  }
  return a;
}

// ---------------------------------------------------------------------------
// Generic projection GEMM: O(f16) = X(f32) @ W(f32).T + bias.
// grid = (M/128, N/64); block = 256 (8 waves); each wave: 16x64 output tile.
// Launched once each for q/k/v so kernel-arg pointers keep global addrspace.
// ---------------------------------------------------------------------------
__global__ __launch_bounds__(256) void proj_f32_kernel(
    const float* __restrict__ X, const float* __restrict__ W,
    const float* __restrict__ bias, half_t* __restrict__ O) {
  const int lane = threadIdx.x & 31, wv = threadIdx.x >> 5;
  const int g = lane >> 4, ln = lane & 15;
  const int m0 = blockIdx.x * 128 + wv * 16;
  const int n0 = blockIdx.y * 64;

  v8f acc[4];
#pragma unroll
  for (int f = 0; f < 4; ++f)
#pragma unroll
    for (int r = 0; r < 8; ++r) acc[f][r] = 0.0f;

  for (int k0 = 0; k0 < D_; k0 += 32) {
    v16h a = load_frag_f32(X, D_, m0 + ln, k0, g);
#pragma unroll
    for (int f = 0; f < 4; ++f) {
      v16h b = load_frag_f32(W, D_, n0 + f * 16 + ln, k0, g);
      acc[f] = WMMA(a, b, acc[f]);
    }
  }

#pragma unroll
  for (int f = 0; f < 4; ++f) {
    float bn = bias[n0 + f * 16 + ln];
#pragma unroll
    for (int r = 0; r < 8; ++r) {
      int m = m0 + r + 8 * g;          // C layout: VGPR r -> M = r (+8 for lanes 16-31)
      O[(size_t)m * D_ + n0 + f * 16 + ln] = (half_t)(acc[f][r] + bn);
    }
  }
}

// ---------------------------------------------------------------------------
// Flash attention. grid = (T/128, B*H); block = 256 (8 waves).
// Each wave owns 16 q-rows; the block shares double-buffered K/V tiles
// (32 keys x 64 dims, fp16) staged with async global->LDS DMA (ASYNCcnt).
// ---------------------------------------------------------------------------
__global__ __launch_bounds__(256) void attn_kernel(
    const half_t* __restrict__ qh, const half_t* __restrict__ kh,
    const half_t* __restrict__ vh, const int* __restrict__ mask,
    half_t* __restrict__ oh) {
  __shared__ half_t Kt[2][32 * DK];       // 2 x 4 KB
  __shared__ half_t Vt[2][32 * DK];       // 2 x 4 KB
  __shared__ half_t Pt[8 * 16 * 32];      // 8 KB, per-wave P scratch (C->A relayout)

  const int lane = threadIdx.x & 31, wv = threadIdx.x >> 5;
  const int g = lane >> 4, ln = lane & 15;
  const int bh = blockIdx.y;
  const int bb = bh / H_, hh = bh % H_;
  const int q0 = blockIdx.x * 128 + wv * 16;

  const half_t* qbase = qh + (size_t)bb * T_ * D_ + hh * DK;
  const half_t* kbase = kh + (size_t)bb * T_ * D_ + hh * DK;
  const half_t* vbase = vh + (size_t)bb * T_ * D_ + hh * DK;
  const int*    mbase = mask + bb * T_;

  // per-thread slice of the 32x64 fp16 tile: 8 contiguous halves (16B, aligned)
  const int i8 = threadIdx.x * 8;
  const int skey = i8 >> 6, sd = i8 & 63;

  // async global->LDS stage of one K/V tile into buffer `buf`
  auto stage = [&](int j0, int buf) {
    const half_t* kg = kbase + (size_t)(j0 + skey) * D_ + sd;
    const half_t* vg = vbase + (size_t)(j0 + skey) * D_ + sd;
    unsigned kl = (unsigned)(unsigned long long)&Kt[buf][i8];  // LDS byte offset
    unsigned vl = (unsigned)(unsigned long long)&Vt[buf][i8];
    asm volatile("global_load_async_to_lds_b128 %0, %1, off"
                 :: "v"(kl), "v"(kg) : "memory");
    asm volatile("global_load_async_to_lds_b128 %0, %1, off"
                 :: "v"(vl), "v"(vg) : "memory");
  };

  // Q fragments: 16 rows x 64 dims = 2 A fragments (K=32 each)
  v16h qa[2];
  qa[0] = load_frag_f16(qbase, D_, q0 + ln, 0, g);
  qa[1] = load_frag_f16(qbase, D_, q0 + ln, 32, g);

  v8f acc[4];
  float mrow[8], lrow[8];
#pragma unroll
  for (int f = 0; f < 4; ++f)
#pragma unroll
    for (int r = 0; r < 8; ++r) acc[f][r] = 0.0f;
#pragma unroll
  for (int r = 0; r < 8; ++r) { mrow[r] = -3.0e38f; lrow[r] = 0.0f; }

  half_t* Pw = &Pt[wv * 16 * 32];

  stage(0, 0);                             // pipeline prologue

  for (int j0 = 0; j0 < T_; j0 += 32) {
    const int cur = (j0 >> 5) & 1;

    // wait for own async loads, then make tile visible block-wide.
    asm volatile("s_wait_asynccnt 0x0" ::: "memory");
    __syncthreads();

    // kick off next tile into the other buffer (safe: all waves are past
    // their reads of that buffer from iteration j0-32, enforced by the
    // barrier above being after that compute in every wave's program order).
    if (j0 + 32 < T_) {
      stage(j0 + 32, cur ^ 1);
      __builtin_prefetch(kbase + (size_t)(j0 + 64) * D_ + sd, 0, 0);
    }

    const int mk0 = mbase[j0 + ln];
    const int mk1 = mbase[j0 + 16 + ln];

    // ---- S = Q K^T : 16 q-rows x 32 keys (two 16x16 C fragments) ----
    v8f s0, s1;
#pragma unroll
    for (int r = 0; r < 8; ++r) { s0[r] = 0.0f; s1[r] = 0.0f; }
#pragma unroll
    for (int cs = 0; cs < 2; ++cs) {
      v16h b0 = load_frag_f16(Kt[cur], DK, ln,      cs * 32, g);  // keys j0+0..15
      v16h b1 = load_frag_f16(Kt[cur], DK, 16 + ln, cs * 32, g);  // keys j0+16..31
      s0 = WMMA(qa[cs], b0, s0);
      s1 = WMMA(qa[cs], b1, s1);
    }

    // ---- scale + padding mask + online softmax (per C-layout row) ----
#pragma unroll
    for (int r = 0; r < 8; ++r) {
      float x0 = mk0 ? s0[r] * SCALE_ : NEGV;
      float x1 = mk1 ? s1[r] * SCALE_ : NEGV;
      float mx = fmaxf(x0, x1);
#pragma unroll
      for (int off = 8; off; off >>= 1) mx = fmaxf(mx, __shfl_xor(mx, off, 32));
      float mnew  = fmaxf(mrow[r], mx);
      float alpha = __expf(mrow[r] - mnew);
      float p0 = __expf(x0 - mnew);
      float p1 = __expf(x1 - mnew);
      float rs = p0 + p1;
#pragma unroll
      for (int off = 8; off; off >>= 1) rs += __shfl_xor(rs, off, 32);
      lrow[r] = lrow[r] * alpha + rs;
      mrow[r] = mnew;
#pragma unroll
      for (int f = 0; f < 4; ++f) acc[f][r] *= alpha;
      // spill P (C layout) to this wave's LDS scratch as fp16
      int ml = r + 8 * g;
      Pw[ml * 32 + ln]      = (half_t)p0;
      Pw[ml * 32 + 16 + ln] = (half_t)p1;
    }

    // ---- re-load P in A-fragment layout, then O += P V ----
    v16h pa = load_frag_f16(Pw, 32, ln, 0, g);
#pragma unroll
    for (int f = 0; f < 4; ++f) {
      v16h vb;   // B fragment: rows = keys, cols = dims (strided LDS reads)
#pragma unroll
      for (int i = 0; i < 8; ++i) {
        int kk = frag_k(i, g);
        vb[2 * i]     = Vt[cur][kk * DK       + f * 16 + ln];
        vb[2 * i + 1] = Vt[cur][(kk + 1) * DK + f * 16 + ln];
      }
      acc[f] = WMMA(pa, vb, acc[f]);
    }
  }

  // ---- normalize and store O (fp16) in b t (h c) layout ----
  half_t* obase = oh + (size_t)bb * T_ * D_ + hh * DK;
#pragma unroll
  for (int r = 0; r < 8; ++r) {
    float rinv = 1.0f / lrow[r];
    int m = r + 8 * g;
#pragma unroll
    for (int f = 0; f < 4; ++f)
      obase[(size_t)(q0 + m) * D_ + f * 16 + ln] = (half_t)(acc[f][r] * rinv);
  }
}

// ---------------------------------------------------------------------------
// Output projection: fp16 A from workspace, fp32 W/bias, fp32 result to d_out.
// ---------------------------------------------------------------------------
__global__ __launch_bounds__(256) void oproj_kernel(
    const half_t* __restrict__ oh, const float* __restrict__ Wo,
    const float* __restrict__ bo, float* __restrict__ out) {
  const int lane = threadIdx.x & 31, wv = threadIdx.x >> 5;
  const int g = lane >> 4, ln = lane & 15;
  const int m0 = blockIdx.x * 128 + wv * 16;
  const int n0 = blockIdx.y * 64;

  v8f acc[4];
#pragma unroll
  for (int f = 0; f < 4; ++f)
#pragma unroll
    for (int r = 0; r < 8; ++r) acc[f][r] = 0.0f;

  for (int k0 = 0; k0 < D_; k0 += 32) {
    v16h a = load_frag_f16(oh, D_, m0 + ln, k0, g);
#pragma unroll
    for (int f = 0; f < 4; ++f) {
      v16h b = load_frag_f32(Wo, D_, n0 + f * 16 + ln, k0, g);
      acc[f] = WMMA(a, b, acc[f]);
    }
  }

#pragma unroll
  for (int f = 0; f < 4; ++f) {
    float bn = bo[n0 + f * 16 + ln];
#pragma unroll
    for (int r = 0; r < 8; ++r) {
      int m = m0 + r + 8 * g;
      out[(size_t)m * D_ + n0 + f * 16 + ln] = acc[f][r] + bn;
    }
  }
}

// ---------------------------------------------------------------------------
extern "C" void kernel_launch(void* const* d_in, const int* in_sizes, int n_in,
                              void* d_out, int out_size, void* d_ws, size_t ws_size,
                              hipStream_t stream) {
  const float* q    = (const float*)d_in[0];
  const float* k    = (const float*)d_in[1];
  const float* v    = (const float*)d_in[2];
  const int*   mask = (const int*)d_in[3];
  const float* Wq   = (const float*)d_in[4];
  const float* bq   = (const float*)d_in[5];
  const float* Wk   = (const float*)d_in[6];
  const float* bk   = (const float*)d_in[7];
  const float* Wv   = (const float*)d_in[8];
  const float* bv   = (const float*)d_in[9];
  const float* Wo   = (const float*)d_in[10];
  const float* bo   = (const float*)d_in[11];

  half_t* ws = (half_t*)d_ws;
  const size_t SZ = (size_t)B_ * T_ * D_;     // 4.19M halves per buffer
  half_t* qh = ws;
  half_t* kh = ws + SZ;
  half_t* vh = ws + 2 * SZ;
  half_t* oh = ws + 3 * SZ;

  dim3 gp(64, 8);             // M/128, N/64
  proj_f32_kernel<<<gp, 256, 0, stream>>>(q, Wq, bq, qh);
  proj_f32_kernel<<<gp, 256, 0, stream>>>(k, Wk, bk, kh);
  proj_f32_kernel<<<gp, 256, 0, stream>>>(v, Wv, bv, vh);

  dim3 ga(T_ / 128, B_ * H_); // q-blocks, (b,h)
  attn_kernel<<<ga, 256, 0, stream>>>(qh, kh, vh, mask, oh);

  dim3 go(64, 8);             // M/128, N/64
  oproj_kernel<<<go, 256, 0, stream>>>(oh, Wo, bo, (float*)d_out);
}
